// QLinear_25615184953514
// MI455X (gfx1250) — 8-bit matrix operand order repaired and checked by reading, compiled, not run on hardware
//
#include <hip/hip_runtime.h>
#include <stdint.h>


typedef int v8i __attribute__((ext_vector_type(8)));

#define M_TOK   16384
#define KDIM    2048
#define NDIM    2048

__global__ __launch_bounds__(256) void qlin_quant_kernel(
    const float* __restrict__ x,
    int8_t* __restrict__ q,
    float* __restrict__ scales)
{
    const int wave = threadIdx.x >> 5;
    const int lane = threadIdx.x & 31;
    const int m    = blockIdx.x * 8 + wave;

    const float4* xr = (const float4*)(x + (size_t)m * KDIM);

    float4 v[16];
    float amax = 0.0f;
#pragma unroll
    for (int c = 0; c < 16; ++c) {
        float4 t = xr[c * 32 + lane];
        v[c] = t;
        amax = fmaxf(amax,
               fmaxf(fmaxf(fabsf(t.x), fabsf(t.y)),
                     fmaxf(fabsf(t.z), fabsf(t.w))));
    }
#pragma unroll
    for (int off = 16; off > 0; off >>= 1)
        amax = fmaxf(amax, __shfl_xor(amax, off, 32));

    const float s   = fmaxf(amax, 1e-7f) * (1.0f / 127.0f);
    const float inv = 1.0f / s;
    if (lane == 0) scales[m] = s;

    int* qo = (int*)(q + (size_t)m * KDIM);
#pragma unroll
    for (int c = 0; c < 16; ++c) {
        float4 t = v[c];
        int b0 = min(127, max(-128, __float2int_rn(t.x * inv)));
        int b1 = min(127, max(-128, __float2int_rn(t.y * inv)));
        int b2 = min(127, max(-128, __float2int_rn(t.z * inv)));
        int b3 = min(127, max(-128, __float2int_rn(t.w * inv)));
        qo[c * 32 + lane] = (b0 & 255) | ((b1 & 255) << 8) |
                            ((b2 & 255) << 16) | (b3 << 24);
    }
}

__global__ __launch_bounds__(128) void qlin_gemm_kernel(
    const int8_t* __restrict__ q,
    const int8_t* __restrict__ w,
    const float*  __restrict__ scales,
    const float*  __restrict__ wscale,
    const float*  __restrict__ bias,
    _Float16*     __restrict__ out)
{
    const int wave  = threadIdx.x >> 5;
    const int lane  = threadIdx.x & 31;
    const int lhalf = lane >> 4;
    const int l15   = lane & 15;

    const int m0 = blockIdx.y * 128 + (wave & 1) * 64;
    const int n0 = blockIdx.x * 128 + (wave >> 1) * 64;

    const v8i vzero = {};
    v8i acc[4][4];
#pragma unroll
    for (int mi = 0; mi < 4; ++mi)
#pragma unroll
        for (int ni = 0; ni < 4; ++ni)
            acc[mi][ni] = vzero;

    const int8_t* arow[4];
    const int8_t* brow[4];
#pragma unroll
    for (int mi = 0; mi < 4; ++mi)
        arow[mi] = q + (size_t)(m0 + mi * 16 + l15) * KDIM + lhalf * 16;
#pragma unroll
    for (int ni = 0; ni < 4; ++ni)
        brow[ni] = w + (size_t)(n0 + ni * 16 + l15) * KDIM + lhalf * 16;

#pragma unroll 2
    for (int k = 0; k < KDIM; k += 64) {
        v8i a[4], b[4];
#pragma unroll
        for (int mi = 0; mi < 4; ++mi) {
            int2 d0 = *(const int2*)(arow[mi] + k);
            int2 d1 = *(const int2*)(arow[mi] + k + 8);
            int2 d2 = *(const int2*)(arow[mi] + k + 32);
            int2 d3 = *(const int2*)(arow[mi] + k + 40);
            v8i t;
            t[0] = d0.x; t[1] = d0.y; t[2] = d1.x; t[3] = d1.y;
            t[4] = d2.x; t[5] = d2.y; t[6] = d3.x; t[7] = d3.y;
            a[mi] = t;
        }
#pragma unroll
        for (int ni = 0; ni < 4; ++ni) {
            int4 e0 = *(const int4*)(brow[ni] + k);
            int4 e1 = *(const int4*)(brow[ni] + k + 32);
            v8i t;
            t[0] = e0.x; t[1] = e0.y; t[2] = e0.z; t[3] = e0.w;
            t[4] = e1.x; t[5] = e1.y; t[6] = e1.z; t[7] = e1.w;
            b[ni] = t;
        }
        if (k + 64 < KDIM) {
            __builtin_prefetch(arow[0] + k + 64, 0, 1);
            __builtin_prefetch(brow[0] + k + 64, 0, 1);
        }
#pragma unroll
        for (int mi = 0; mi < 4; ++mi)
#pragma unroll
            for (int ni = 0; ni < 4; ++ni)
                acc[mi][ni] = __builtin_amdgcn_wmma_i32_16x16x64_iu8(
                     true,  a[mi],
                     true,  b[ni],
                    acc[mi][ni],
                     false,  false);
    }

    float wsn[4], bn[4];
#pragma unroll
    for (int ni = 0; ni < 4; ++ni) {
        const int n = n0 + ni * 16 + l15;
        wsn[ni] = wscale[n];
        bn[ni]  = bias[n];
    }
#pragma unroll
    for (int mi = 0; mi < 4; ++mi) {
        const int mbase = m0 + mi * 16 + lhalf * 8;
        float sm[8];
#pragma unroll
        for (int i = 0; i < 8; ++i) sm[i] = scales[mbase + i];
#pragma unroll
        for (int ni = 0; ni < 4; ++ni) {
            const int n = n0 + ni * 16 + l15;
#pragma unroll
            for (int i = 0; i < 8; ++i) {
                const int m = mbase + i;
                float r = (float)acc[mi][ni][i] * sm[i] * wsn[ni] + bn[ni];
                out[(size_t)m * NDIM + n] = (_Float16)r;
            }
        }
    }
}

extern "C" void kernel_launch(void* const* d_in, const int* in_sizes, int n_in,
                              void* d_out, int out_size, void* d_ws, size_t ws_size,
                              hipStream_t stream)
{
    const float*  x      = (const float*)d_in[0];
    const int8_t* weight = (const int8_t*)d_in[1];
    const float*  wscale = (const float*)d_in[2];
    const float*  bias   = (const float*)d_in[3];
    _Float16*     out    = (_Float16*)d_out;

    int8_t* q      = (int8_t*)d_ws;
    float*  scales = (float*)((char*)d_ws + (size_t)M_TOK * KDIM);

    qlin_quant_kernel<<<M_TOK / 8, 256, 0, stream>>>(x, q, scales);

    dim3 grid(NDIM / 128, M_TOK / 128);
    qlin_gemm_kernel<<<grid, 128, 0, stream>>>(q, weight, scales, wscale, bias, out);
}
